// VQVAE_5652176961857
// MI455X (gfx1250) — matrix operand order repaired and checked by reading, not compiled, not run on hardware
//
#include <hip/hip_runtime.h>
#include <cstddef>

// ---------------------------------------------------------------------------
// Types for CDNA5 WMMA (wave32)
// ---------------------------------------------------------------------------
typedef _Float16 h4_t   __attribute__((ext_vector_type(4)));
typedef _Float16 h8_t   __attribute__((ext_vector_type(8)));
typedef _Float16 v16h_t __attribute__((ext_vector_type(16)));
typedef float    v8f_t  __attribute__((ext_vector_type(8)));

union FragH { v16h_t v; h8_t h[2]; };
union FragE { v16h_t v; _Float16 e[16]; };

#define NBATCH 32
#define TFULL  2048
#define INDIM  263
#define WID    512
#define TD     512
#define NCODE  1024

#define KC 32
#define ASTR 40    // LDS row stride in halves (80B: keeps b64/b128 alignment, spreads banks)
#define AFSTR 520

// ---------------------------------------------------------------------------
// Generic conv1d as implicit GEMM on v_wmma_f32_16x16x32_f16.
// Activations are f16 end-to-end (halves HBM traffic; 512x2048 tensors become
// L2-resident at 64MB).  Weights/bias stay f32 and are converted in staging.
//   X : [N, Cin, Tin] f16   W : [Cout, Cin, Kw] f32   Bias : [Cout] f32
//   Y : [N, Cout, Tout] f16 Res: optional residual [N, Cout, Tout] f16
// Block = 256 threads = 8 waves. Block tile = 64 co x 128 t.
// Wave tile = 32 co x 32 t = 2x2 WMMA accumulators (4 v_wmma per K-chunk).
// ALIGNED=1: Cin%32==0 && Cout%64==0 && Tout%128==0  -> no guards anywhere.
// upsample: input is logically repeat(X,2) along time.
// ---------------------------------------------------------------------------
template<int ALIGNED>
__global__ __launch_bounds__(256)
void conv1d_wmma_kernel(const _Float16* __restrict__ X, const float* __restrict__ W,
                        const float* __restrict__ Bias, const _Float16* __restrict__ Res,
                        _Float16* __restrict__ Y,
                        int Cin, int Cout, int Tin, int Tout,
                        int Kw, int stride, int pad, int dil,
                        int reluIn, int reluOut, int upsample)
{
    __shared__ __align__(16) _Float16 As[64][ASTR];    // [co][k]
    __shared__ __align__(16) _Float16 Bs[128][ASTR];   // [t][k]

    const int tid  = threadIdx.x;
    const int lane = tid & 31;
    const int wave = tid >> 5;
    const int chSub = wave & 1;          // 2 x 32-co sub-tiles
    const int tSub  = wave >> 1;         // 4 x 32-t  sub-tiles
    const int rsel  = lane & 15;
    const int half  = lane >> 4;

    const int n      = blockIdx.z;
    const int coBase = blockIdx.y * 64;
    const int tBase  = blockIdx.x * 128;

    const _Float16* Xn = X + (size_t)n * Cin * Tin;
    const int TinEff = upsample ? (Tin << 1) : Tin;

    // staging coordinates (fixed per thread)
    const int aKq  = (tid & 7) * 4;       // A: k quad within row
    const int aCo0 = (tid >> 3);          // A: row for i=0 (i=1 -> +32)
    const int bT   = tid & 127;           // B: row (time), same for all i
    const int bKq0 = (tid >> 7);          // B: k quad index base (i -> +2i)

    v8f_t acc[2][2];
    #pragma unroll
    for (int mi = 0; mi < 2; ++mi)
        #pragma unroll
        for (int ni = 0; ni < 2; ++ni)
            acc[mi][ni] = v8f_t{0.f,0.f,0.f,0.f,0.f,0.f,0.f,0.f};

    for (int kt = 0; kt < Kw; ++kt) {
        // time index for this thread's B row, hoisted out of the channel loop
        int tin = (tBase + bT) * stride - pad + kt * dil;
        const bool binb = (tin >= 0) && (tin < TinEff);
        int ts = upsample ? (tin >> 1) : tin;
        ts = min(max(ts, 0), Tin - 1);                 // clamped -> always safe to load

        for (int ci0 = 0; ci0 < Cin; ci0 += KC) {
            __syncthreads();             // protect previous iteration's fragment reads

            // ---- stage A: weights (f32 -> f16) -> As (64x32), branchless
            #pragma unroll
            for (int i = 0; i < 2; ++i) {
                const int co  = aCo0 + 32 * i;
                const int gco = coBase + co;
                h4_t q;
                #pragma unroll
                for (int j = 0; j < 4; ++j) {
                    const int gci = ci0 + aKq + j;
                    float w;
                    if (ALIGNED) {
                        w = W[((size_t)gco * Cin + gci) * Kw + kt];
                    } else {
                        const int cco = min(gco, Cout - 1);
                        const int cci = min(gci, Cin - 1);
                        w = W[((size_t)cco * Cin + cci) * Kw + kt];
                        w = (gco < Cout && gci < Cin) ? w : 0.f;
                    }
                    q[j] = (_Float16)w;
                }
                *(h4_t*)&As[co][aKq] = q;
            }

            // ---- stage B: f16 inputs -> Bs (128x32), branchless, no cvt
            #pragma unroll
            for (int i = 0; i < 4; ++i) {
                const int kq = (bKq0 + 2 * i) * 4;
                h4_t q;
                #pragma unroll
                for (int j = 0; j < 4; ++j) {
                    const int gci = ci0 + kq + j;
                    _Float16 v;
                    if (ALIGNED) {
                        v = Xn[(size_t)gci * Tin + ts];
                        v = binb ? v : (_Float16)0.f;
                    } else {
                        const int cci = min(gci, Cin - 1);
                        v = Xn[(size_t)cci * Tin + ts];
                        v = (binb && gci < Cin) ? v : (_Float16)0.f;
                    }
                    if (reluIn) v = (v > (_Float16)0.f) ? v : (_Float16)0.f;
                    q[j] = v;
                }
                *(h4_t*)&Bs[bT][kq] = q;
            }

            if (ci0 + KC < Cin)
                __builtin_prefetch(&Xn[(size_t)(ci0 + KC) * Tin + ts], 0, 1);
            __syncthreads();

            // ---- fragments (ISA wave32 layouts) + 4 WMMAs
            FragH a0, a1, b0, b1;
            {
                const h8_t* r0 = (const h8_t*)&As[chSub * 32 + rsel][0];
                const h8_t* r1 = (const h8_t*)&As[chSub * 32 + 16 + rsel][0];
                a0.h[0] = r0[half]; a0.h[1] = r0[half + 2];
                a1.h[0] = r1[half]; a1.h[1] = r1[half + 2];
            }
            {
                const h8_t* r0 = (const h8_t*)&Bs[tSub * 32 + rsel][0];
                const h8_t* r1 = (const h8_t*)&Bs[tSub * 32 + 16 + rsel][0];
                b0.h[0] = r0[half]; b0.h[1] = r0[half + 2];
                b1.h[0] = r1[half]; b1.h[1] = r1[half + 2];
            }
            acc[0][0] = __builtin_amdgcn_wmma_f32_16x16x32_f16(false, a0.v, false, b0.v, (short)0, acc[0][0], false, false);
            acc[0][1] = __builtin_amdgcn_wmma_f32_16x16x32_f16(false, a0.v, false, b1.v, (short)0, acc[0][1], false, false);
            acc[1][0] = __builtin_amdgcn_wmma_f32_16x16x32_f16(false, a1.v, false, b0.v, (short)0, acc[1][0], false, false);
            acc[1][1] = __builtin_amdgcn_wmma_f32_16x16x32_f16(false, a1.v, false, b1.v, (short)0, acc[1][1], false, false);
        }
    }

    // ---- epilogue: D layout — elem v: M = v + 8*half, N = lane%16
    #pragma unroll
    for (int mi = 0; mi < 2; ++mi) {
        const int coOff = coBase + chSub * 32 + mi * 16 + half * 8;
        #pragma unroll
        for (int ni = 0; ni < 2; ++ni) {
            const int tOut = tBase + tSub * 32 + ni * 16 + rsel;
            #pragma unroll
            for (int v = 0; v < 8; ++v) {
                const int co = coOff + v;
                if (ALIGNED || (co < Cout && tOut < Tout)) {
                    float val = acc[mi][ni][v] + Bias[co];
                    if (Res) val += (float)Res[((size_t)n * Cout + co) * Tout + tOut];
                    if (reluOut) val = fmaxf(val, 0.f);
                    Y[((size_t)n * Cout + co) * Tout + tOut] = (_Float16)val;
                }
            }
        }
    }
}

// ---------------------------------------------------------------------------
// Layout helpers
// ---------------------------------------------------------------------------
__global__ void transpose_in_kernel(const float* __restrict__ X, _Float16* __restrict__ Y,
                                    int C, int Tt, size_t tot)
{
    size_t i = (size_t)blockIdx.x * 256 + threadIdx.x;
    if (i >= tot) return;
    int t = (int)(i % Tt);
    size_t nc = i / Tt;
    int c = (int)(nc % C);
    int n = (int)(nc / C);
    Y[i] = (_Float16)X[((size_t)n * Tt + t) * C + c];
}

__global__ void transpose_out_kernel(const _Float16* __restrict__ X, const float* __restrict__ mask,
                                     float* __restrict__ Y, int C, int Tt, size_t tot)
{
    size_t i = (size_t)blockIdx.x * 256 + threadIdx.x;
    if (i >= tot) return;
    int c = (int)(i % C);
    size_t nt = i / C;
    int t = (int)(nt % Tt);
    int n = (int)(nt / Tt);
    Y[i] = (float)X[((size_t)n * C + c) * Tt + t] * mask[(size_t)n * Tt + t];
}

__global__ void cb_norm_kernel(const float* __restrict__ CB, float* __restrict__ cn,
                               int nCodes, int C)
{
    int j = blockIdx.x * blockDim.x + threadIdx.x;
    if (j >= nCodes) return;
    float s = 0.f;
    for (int c = 0; c < C; ++c) { float v = CB[(size_t)j * C + c]; s += v * v; }
    cn[j] = s;
}

// ---------------------------------------------------------------------------
// VQ argmin via WMMA: scores s = f . c ; metric m = ||c||^2 - 2 s.
// Block = 16 rows (flat (n,td) vectors), 8 waves each scan 8 code-tiles of 16.
// ---------------------------------------------------------------------------
__global__ __launch_bounds__(256)
void vq_argmin_kernel(const _Float16* __restrict__ XE, const float* __restrict__ CB,
                      const float* __restrict__ cn, const float* __restrict__ mask,
                      int* __restrict__ idxOut, float* __restrict__ idxOutF,
                      int* __restrict__ hist,
                      int Cdim, int Td, int Tfull, int nCodes)
{
    __shared__ __align__(16) _Float16 Af[16][AFSTR];
    __shared__ float redM[16][128];
    __shared__ int   redJ[16][128];

    const int tid = threadIdx.x, lane = tid & 31, wave = tid >> 5;
    const int rsel = lane & 15, half = lane >> 4;
    const int rowBase = blockIdx.x * 16;
    const int n   = rowBase / Td;
    const int td0 = rowBase % Td;

    #pragma unroll
    for (int i = 0; i < 32; ++i) {
        int idx = tid + i * 256;
        int c = idx >> 4;
        int r = idx & 15;
        int td = td0 + r;
        float dm = mask[(size_t)n * Tfull + 4 * td];
        Af[r][c] = (_Float16)((float)XE[((size_t)n * Cdim + c) * Td + td] * dm);
    }
    __syncthreads();

    float bestM[8]; int bestJ[8];
    #pragma unroll
    for (int v = 0; v < 8; ++v) { bestM[v] = 3.4e38f; bestJ[v] = 0x7fffffff; }

    for (int jt = wave; jt < nCodes / 16; jt += 8) {
        const int j = jt * 16 + rsel;
        v8f_t s = {0.f,0.f,0.f,0.f,0.f,0.f,0.f,0.f};
        for (int kc = 0; kc < Cdim; kc += KC) {
            FragH a; FragE b;
            const h8_t* arow = (const h8_t*)&Af[rsel][kc];
            a.h[0] = arow[half * 2];
            a.h[1] = arow[half * 2 + 1];
            const float* cp = &CB[(size_t)j * Cdim + kc + 16 * half];
            #pragma unroll
            for (int e = 0; e < 16; ++e) b.e[e] = (_Float16)cp[e];
            s = __builtin_amdgcn_wmma_f32_16x16x32_f16(
                    false, a.v, false, b.v, (short)0, s, false, false);
        }
        float cnj = cn[j];
        #pragma unroll
        for (int v = 0; v < 8; ++v) {
            float m = cnj - 2.0f * s[v];
            if (m < bestM[v] || (m == bestM[v] && j < bestJ[v])) { bestM[v] = m; bestJ[v] = j; }
        }
    }
    const int slot = rsel + 16 * wave;
    #pragma unroll
    for (int v = 0; v < 8; ++v) {
        int r = v + 8 * half;
        redM[r][slot] = bestM[v];
        redJ[r][slot] = bestJ[v];
    }
    __syncthreads();
    if (tid < 16) {
        float bm = 3.4e38f; int bj = 0x7fffffff;
        for (int s2 = 0; s2 < 128; ++s2) {
            float m = redM[tid][s2]; int j = redJ[tid][s2];
            if (m < bm || (m == bm && j < bj)) { bm = m; bj = j; }
        }
        int row = rowBase + tid;
        idxOut[row]  = bj;
        idxOutF[row] = (float)bj;
        atomicAdd(&hist[bj], 1);
    }
}

// gather xq = codebook[idx]; commit accumulation; f16 xq for decoder, f32 for d_out
__global__ __launch_bounds__(256)
void vq_gather_kernel(const _Float16* __restrict__ XE, const float* __restrict__ CB,
                      const float* __restrict__ mask, const int* __restrict__ idx,
                      _Float16* __restrict__ xqBuf, float* __restrict__ xqOut,
                      float* __restrict__ commitAcc, int Cdim, int Td, int Tfull)
{
    __shared__ float red[256];
    const int row = blockIdx.x;            // n*Td + td
    const int n = row / Td, td = row % Td;
    const int j = idx[row];
    const float dm = mask[(size_t)n * Tfull + 4 * td];
    float local = 0.f;
    for (int c = threadIdx.x; c < Cdim; c += 256) {
        float q = CB[(size_t)j * Cdim + c];
        float f = (float)XE[((size_t)n * Cdim + c) * Td + td] * dm;
        float d = f - q;
        local += d * d;
        float xm = q * dm;
        xqBuf[((size_t)n * Cdim + c) * Td + td] = (_Float16)xm;
        xqOut[(size_t)row * Cdim + c] = xm;
    }
    red[threadIdx.x] = local;
    __syncthreads();
    for (int s = 128; s > 0; s >>= 1) {
        if (threadIdx.x < s) red[threadIdx.x] += red[threadIdx.x + s];
        __syncthreads();
    }
    if (threadIdx.x == 0) atomicAdd(commitAcc, red[0]);
}

__global__ void vq_finalize_kernel(const int* __restrict__ hist, const float* __restrict__ commitAcc,
                                   float* __restrict__ outCommit, float* __restrict__ outPerp,
                                   int nCodes, float invTotal, float invCommitN)
{
    __shared__ float red[256];
    float s = 0.f;
    for (int j = threadIdx.x; j < nCodes; j += 256) {
        float p = (float)hist[j] * invTotal;
        s += p * logf(p + 1e-7f);
    }
    red[threadIdx.x] = s;
    __syncthreads();
    for (int st = 128; st > 0; st >>= 1) {
        if (threadIdx.x < st) red[threadIdx.x] += red[threadIdx.x + st];
        __syncthreads();
    }
    if (threadIdx.x == 0) {
        *outPerp   = expf(-red[0]);
        *outCommit = *commitAcc * invCommitN;
    }
}

// ---------------------------------------------------------------------------
// Host side: parameter mapping + launch sequence
// ---------------------------------------------------------------------------
struct ConvW { const float* w; const float* b; };
struct Res3  { ConvW c1, c2; };
struct NetP {
    const float *x, *mask, *codebook;
    ConvW enc_in, enc_down[2], enc_out;
    Res3  enc_res[2][3];
    ConvW dec_in, dec_up[2], dec_out1, dec_out2;
    Res3  dec_res[2][3];
};

static inline const float* F(void* const* d, int i) { return (const float*)d[i]; }

// insertion-order flattening of setup_inputs()
static void map_insertion(void* const* d, NetP& P) {
    int i = 0;
    P.x = F(d,i++); P.mask = F(d,i++);
    P.enc_in.w = F(d,i++); P.enc_in.b = F(d,i++);
    for (int b = 0; b < 2; ++b) {
        P.enc_down[b].w = F(d,i++); P.enc_down[b].b = F(d,i++);
        for (int r = 0; r < 3; ++r) {
            P.enc_res[b][r].c1.w = F(d,i++); P.enc_res[b][r].c1.b = F(d,i++);
            P.enc_res[b][r].c2.w = F(d,i++); P.enc_res[b][r].c2.b = F(d,i++);
        }
    }
    P.enc_out.w = F(d,i++); P.enc_out.b = F(d,i++);
    P.dec_in.w = F(d,i++); P.dec_in.b = F(d,i++);
    for (int b = 0; b < 2; ++b) {
        for (int r = 0; r < 3; ++r) {
            P.dec_res[b][r].c1.w = F(d,i++); P.dec_res[b][r].c1.b = F(d,i++);
            P.dec_res[b][r].c2.w = F(d,i++); P.dec_res[b][r].c2.b = F(d,i++);
        }
        P.dec_up[b].w = F(d,i++); P.dec_up[b].b = F(d,i++);
    }
    P.dec_out1.w = F(d,i++); P.dec_out1.b = F(d,i++);
    P.dec_out2.w = F(d,i++); P.dec_out2.b = F(d,i++);
    P.codebook = F(d,i++);
}

// jax-pytree (alphabetical dict keys) flattening
static void map_alpha(void* const* d, NetP& P) {
    int i = 0;
    P.codebook = F(d,i++);
    for (int b = 0; b < 2; ++b) {
        for (int r = 0; r < 3; ++r) {
            P.dec_res[b][r].c1.b = F(d,i++);
            P.dec_res[b][r].c2.b = F(d,i++);
            P.dec_res[b][r].c1.w = F(d,i++);
            P.dec_res[b][r].c2.w = F(d,i++);
        }
        P.dec_up[b].b = F(d,i++); P.dec_up[b].w = F(d,i++);
    }
    P.dec_in.b = F(d,i++); P.dec_in.w = F(d,i++);
    P.dec_out1.b = F(d,i++); P.dec_out1.w = F(d,i++);
    P.dec_out2.b = F(d,i++); P.dec_out2.w = F(d,i++);
    for (int b = 0; b < 2; ++b) {
        P.enc_down[b].b = F(d,i++); P.enc_down[b].w = F(d,i++);
        for (int r = 0; r < 3; ++r) {
            P.enc_res[b][r].c1.b = F(d,i++);
            P.enc_res[b][r].c2.b = F(d,i++);
            P.enc_res[b][r].c1.w = F(d,i++);
            P.enc_res[b][r].c2.w = F(d,i++);
        }
    }
    P.enc_in.b = F(d,i++); P.enc_in.w = F(d,i++);
    P.enc_out.b = F(d,i++); P.enc_out.w = F(d,i++);
    P.mask = F(d,i++); P.x = F(d,i++);
}

extern "C" void kernel_launch(void* const* d_in, const int* in_sizes, int n_in,
                              void* d_out, int out_size, void* d_ws, size_t ws_size,
                              hipStream_t stream)
{
    (void)n_in; (void)out_size; (void)ws_size;
    NetP P;
    if (in_sizes[0] == NCODE * WID) map_alpha(d_in, P);
    else                            map_insertion(d_in, P);

    const size_t BUFN = (size_t)NBATCH * WID * TFULL;   // halves per buffer
    _Float16* A = (_Float16*)d_ws;
    _Float16* B = A + BUFN;
    _Float16* C = B + BUFN;
    char*  S = (char*)(C + BUFN);
    float* cn     = (float*)S;                       // 1024 f32
    int*   idxI   = (int*)(S + 4096);                // 16384 i32
    int*   hist   = (int*)(S + 4096 + 65536);        // 1024 i32
    float* commit = (float*)(S + 4096 + 65536 + 4096);

    float* out0      = (float*)d_out;                                   // [N,T,263]
    float* outIdx    = out0 + (size_t)NBATCH * TFULL * INDIM;           // [N,Td] as f32
    float* outCommit = outIdx + (size_t)NBATCH * TD;
    float* outPerp   = outCommit + 1;
    float* outXq     = outPerp + 1;                                     // [N,Td,C]

    hipMemsetAsync(hist, 0, 4096 + 4, stream);   // hist + commit accumulator

    auto CONV = [&](const _Float16* Xp, const ConvW& cw, const _Float16* Rp, _Float16* Yp,
                    int Cin, int Cout, int Tin, int Tout,
                    int Kw, int st, int pd, int dl, int ri, int ro, int up) {
        dim3 g((Tout + 127) / 128, (Cout + 63) / 64, NBATCH);
        const bool aligned = (Cin % KC == 0) && (Cout % 64 == 0) && (Tout % 128 == 0);
        if (aligned)
            conv1d_wmma_kernel<1><<<g, 256, 0, stream>>>(Xp, cw.w, cw.b, Rp, Yp,
                Cin, Cout, Tin, Tout, Kw, st, pd, dl, ri, ro, up);
        else
            conv1d_wmma_kernel<0><<<g, 256, 0, stream>>>(Xp, cw.w, cw.b, Rp, Yp,
                Cin, Cout, Tin, Tout, Kw, st, pd, dl, ri, ro, up);
    };

    // x_in = transpose(x) -> A [N,263,2048] f16
    {
        size_t tot = (size_t)NBATCH * INDIM * TFULL;
        transpose_in_kernel<<<(unsigned)((tot + 255) / 256), 256, 0, stream>>>(P.x, A, INDIM, TFULL, tot);
    }

    // ---------------- Encoder ----------------
    CONV(A, P.enc_in, nullptr, B, INDIM, WID, 2048, 2048, 3,1,1,1, 0,1,0);   // relu(conv_in)

    _Float16 *xb, *t1, *t2;
    // block 0: down (2048 -> 1024) then 3 res (dil 1,3,9)
    CONV(B, P.enc_down[0], nullptr, A, WID, WID, 2048, 1024, 4,2,1,1, 0,0,0);
    xb = A; t1 = B; t2 = C;
    {
        int dils[3] = {1, 3, 9};
        for (int r = 0; r < 3; ++r) {
            CONV(xb, P.enc_res[0][r].c1, nullptr, t1, WID, WID, 1024, 1024, 3,1,dils[r],dils[r], 1,0,0);
            CONV(t1, P.enc_res[0][r].c2, xb,      t2, WID, WID, 1024, 1024, 1,1,0,1, 1,0,0);
            _Float16* old = xb; xb = t2; t2 = old;
        }
    }
    // block 1: down (1024 -> 512) then 3 res
    CONV(xb, P.enc_down[1], nullptr, t2, WID, WID, 1024, 512, 4,2,1,1, 0,0,0);
    { _Float16* nx = t2; t2 = xb; xb = nx; }
    {
        int dils[3] = {1, 3, 9};
        for (int r = 0; r < 3; ++r) {
            CONV(xb, P.enc_res[1][r].c1, nullptr, t1, WID, WID, 512, 512, 3,1,dils[r],dils[r], 1,0,0);
            CONV(t1, P.enc_res[1][r].c2, xb,      t2, WID, WID, 512, 512, 1,1,0,1, 1,0,0);
            _Float16* old = xb; xb = t2; t2 = old;
        }
    }
    // enc out conv -> xe
    _Float16* xe = t2;
    CONV(xb, P.enc_out, nullptr, xe, WID, WID, 512, 512, 3,1,1,1, 0,0,0);
    { _Float16* old = xb; xb = xe; t2 = old; }   // xb = xe

    // ---------------- Quantize ----------------
    cb_norm_kernel<<<(NCODE + 255) / 256, 256, 0, stream>>>(P.codebook, cn, NCODE, WID);
    vq_argmin_kernel<<<(NBATCH * TD) / 16, 256, 0, stream>>>(
        xb, P.codebook, cn, P.mask, idxI, outIdx, hist, WID, TD, TFULL, NCODE);
    _Float16* xq = t1;   // xq buffer [N,512,512] f16
    vq_gather_kernel<<<NBATCH * TD, 256, 0, stream>>>(
        xb, P.codebook, P.mask, idxI, xq, outXq, commit, WID, TD, TFULL);
    vq_finalize_kernel<<<1, 256, 0, stream>>>(
        hist, commit, outCommit, outPerp, NCODE,
        1.0f / (float)(NBATCH * TD), 1.0f / ((float)(NBATCH * TD) * (float)WID));

    // ---------------- Decoder ----------------
    _Float16* dx = t2;
    CONV(xq, P.dec_in, nullptr, dx, WID, WID, 512, 512, 3,1,1,1, 0,1,0);     // relu(conv_in)
    xb = dx; t1 = xq;
    t2 = (xb == A) ? ((t1 == B) ? C : B) : ((xb == B) ? ((t1 == A) ? C : A) : ((t1 == A) ? B : A));
    // block 0: res dil 9,3,1 at T=512, then repeat x2 + up conv -> 1024
    {
        int dils[3] = {9, 3, 1};
        for (int r = 0; r < 3; ++r) {
            CONV(xb, P.dec_res[0][r].c1, nullptr, t1, WID, WID, 512, 512, 3,1,dils[r],dils[r], 1,0,0);
            CONV(t1, P.dec_res[0][r].c2, xb,      t2, WID, WID, 512, 512, 1,1,0,1, 1,0,0);
            _Float16* old = xb; xb = t2; t2 = old;
        }
        CONV(xb, P.dec_up[0], nullptr, t2, WID, WID, 512, 1024, 3,1,1,1, 0,0,1);
        _Float16* old = xb; xb = t2; t2 = old;
    }
    // block 1: res dil 9,3,1 at T=1024, then repeat x2 + up conv -> 2048
    {
        int dils[3] = {9, 3, 1};
        for (int r = 0; r < 3; ++r) {
            CONV(xb, P.dec_res[1][r].c1, nullptr, t1, WID, WID, 1024, 1024, 3,1,dils[r],dils[r], 1,0,0);
            CONV(t1, P.dec_res[1][r].c2, xb,      t2, WID, WID, 1024, 1024, 1,1,0,1, 1,0,0);
            _Float16* old = xb; xb = t2; t2 = old;
        }
        CONV(xb, P.dec_up[1], nullptr, t2, WID, WID, 1024, 2048, 3,1,1,1, 0,0,1);
        _Float16* old = xb; xb = t2; t2 = old;
    }
    // out1 (relu) then out2 -> [N,263,2048]
    CONV(xb, P.dec_out1, nullptr, t1, WID, WID, 2048, 2048, 3,1,1,1, 0,1,0);
    CONV(t1, P.dec_out2, nullptr, t2, WID, INDIM, 2048, 2048, 3,1,1,1, 0,0,0);

    // masked transpose to output [N,T,263] f32
    {
        size_t tot = (size_t)NBATCH * TFULL * INDIM;
        transpose_out_kernel<<<(unsigned)((tot + 255) / 256), 256, 0, stream>>>(
            t2, P.mask, out0, INDIM, TFULL, tot);
    }
}
